// AttUnitBiLi_53480932770638
// MI455X (gfx1250) — matrix operand order repaired and checked by reading, not compiled, not run on hardware
//
#include <hip/hip_runtime.h>
#include <hip/hip_bf16.h>
#include <math.h>

// ---------------------------------------------------------------------------
// AttUnitBiLi on MI455X (gfx1250, wave32, WMMA, async-to-LDS):
//   k0      : keys f32 -> bf16 in two layouts (keysN [b][k][d], keysT [b][d][k])
//   GEMM1   : t_query = queries @ W_in^T     (bf16 WMMA, out bf16 -> ws)
//   GEMM2   : alpha = t_query @ keys^T, mask (async-to-LDS staging, out f32)
//   stats   : rowmax, 1/rowsum per (s,b) row
//   GEMM3   : attened = softmax(alpha)@keys  (async-to-LDS B staging, out f32)
// ---------------------------------------------------------------------------

typedef unsigned short ushort_t;
typedef __attribute__((ext_vector_type(16))) __bf16 v16bf;
typedef __attribute__((ext_vector_type(8)))  float  v8f;
typedef __attribute__((ext_vector_type(4)))  int    v4i;

#define S_DIM 1024
#define B_DIM 16
#define K_DIM 1024
#define D_DIM 1024
#define M1    (S_DIM * B_DIM)   // 16384 fused (s,b) rows

#define BM 128
#define BN 128
#define BK 32
#define NKC 32                  // 1024 / BK reduction chunks
#define LDS_PAD 8
#define LDS_STRIDE (BK + LDS_PAD)   // 40 halves = 80 bytes (multiple of 16)

// ---- f32 -> bf16 (prefer v_cvt_pk_bf16_f32) -------------------------------
__device__ __forceinline__ ushort_t f2bf(float x) {
  unsigned u = __builtin_bit_cast(unsigned, x);
  unsigned r = u + 0x7FFFu + ((u >> 16) & 1u);
  return (ushort_t)(r >> 16);
}
__device__ __forceinline__ unsigned pack2(float a, float b) {
#if __has_builtin(__builtin_amdgcn_cvt_pk_bf16_f32)
  typedef __attribute__((ext_vector_type(2))) __bf16 v2bf;
  v2bf r = __builtin_amdgcn_cvt_pk_bf16_f32(a, b);
  return __builtin_bit_cast(unsigned, r);
#else
  return (unsigned)f2bf(a) | ((unsigned)f2bf(b) << 16);
#endif
}

// ---- async global->LDS 16B copy (ASYNCcnt path), sync fallback ------------
#define HAS_ASYNC_LDS __has_builtin(__builtin_amdgcn_global_load_async_to_lds_b128)

__device__ __forceinline__ void async_copy16(const void* g, void* l) {
#if HAS_ASYNC_LDS
  // builtin signature (from hipcc diagnostic): param0 = v4i AS1* (global),
  // param1 = LDS side, then imm offset, imm cpol.
  __builtin_amdgcn_global_load_async_to_lds_b128(
      (__attribute__((address_space(1))) v4i*)const_cast<void*>(g),
      (__attribute__((address_space(3))) v4i*)l,
      0, 0);
#else
  *(uint4*)l = *(const uint4*)g;
#endif
}
__device__ __forceinline__ void async_wait() {
#if HAS_ASYNC_LDS
#if __has_builtin(__builtin_amdgcn_s_wait_asynccnt)
  __builtin_amdgcn_s_wait_asynccnt(0);
#else
  asm volatile("s_wait_asynccnt 0x0" ::: "memory");
#endif
#endif
}

// ---- WMMA fragment loads (ISA 7.12.2 bf16 layouts) ------------------------
union FragU { uint4 q[2]; v16bf v; };

// A 16x32: lane<16 -> M=lane, K in {0..7,16..23}; lane>=16 -> K in {8..15,24..31}
__device__ __forceinline__ v16bf load_fragA(const ushort_t* rowp, int lane) {
  const int r0 = (lane >> 4) * 8;
  FragU f;
  f.q[0] = *(const uint4*)(rowp + r0);
  f.q[1] = *(const uint4*)(rowp + r0 + 16);
  return f.v;
}
// B 32x16: lane<16 -> N=lane, K in {0..7,16..23}; lane>=16 -> K in {8..15,24..31}
__device__ __forceinline__ v16bf load_fragB(const ushort_t* rowp, int lane) {
  const int r0 = (lane >> 4) * 8;
  FragU f;
  f.q[0] = *(const uint4*)(rowp + r0);
  f.q[1] = *(const uint4*)(rowp + r0 + 16);
  return f.v;
}

#define WMMA_BF16(A, Bv, C) \
  __builtin_amdgcn_wmma_f32_16x16x32_bf16(false, (A), false, (Bv), (short)0, (C), false, false)

// 8 WMMAs: one wave computes a 32x64 patch (2 M-subtiles x 4 N-subtiles)
__device__ __forceinline__ void mma_tiles(const ushort_t (*As)[LDS_STRIDE],
                                          const ushort_t (*Bs)[LDS_STRIDE],
                                          int wm, int wn, int lane, v8f acc[2][4]) {
  v16bf aF[2], bF[4];
#pragma unroll
  for (int mt = 0; mt < 2; ++mt)
    aF[mt] = load_fragA(&As[wm * 32 + mt * 16 + (lane & 15)][0], lane);
#pragma unroll
  for (int nt = 0; nt < 4; ++nt)
    bF[nt] = load_fragB(&Bs[wn * 64 + nt * 16 + (lane & 15)][0], lane);
#pragma unroll
  for (int mt = 0; mt < 2; ++mt)
#pragma unroll
    for (int nt = 0; nt < 4; ++nt)
      acc[mt][nt] = WMMA_BF16(aF[mt], bF[nt], acc[mt][nt]);
}

// ------------------- k0: keys f32 -> bf16, normal + transposed --------------
__global__ __launch_bounds__(256)
void k0_convert(const float* __restrict__ Keys, ushort_t* __restrict__ keysN,
                ushort_t* __restrict__ keysT) {
  __shared__ ushort_t tile[32][33];
  const int b = blockIdx.z;
  const int k0 = blockIdx.y * 32, d0 = blockIdx.x * 32;
  const int t = threadIdx.x;
  {
    const int row = t >> 3, col = (t & 7) * 4;        // 32x32 f32 tile, 1 float4/thr
    const float4 v = *(const float4*)(Keys + ((size_t)b * K_DIM + k0 + row) * D_DIM + d0 + col);
    const uint2 p = make_uint2(pack2(v.x, v.y), pack2(v.z, v.w));
    *(uint2*)(keysN + ((size_t)b * K_DIM + k0 + row) * D_DIM + d0 + col) = p;
    *(uint2*)&tile[row][col] = p;
  }
  __syncthreads();
  {
    const int drow = t >> 3, kcol = (t & 7) * 4;      // transposed write, k-contig
    uint2 p;
    ushort_t s0 = tile[kcol + 0][drow], s1 = tile[kcol + 1][drow];
    ushort_t s2 = tile[kcol + 2][drow], s3 = tile[kcol + 3][drow];
    p.x = (unsigned)s0 | ((unsigned)s1 << 16);
    p.y = (unsigned)s2 | ((unsigned)s3 << 16);
    *(uint2*)(keysT + ((size_t)b * D_DIM + d0 + drow) * K_DIM + k0 + kcol) = p;
  }
}

// ------------------------- GEMM1: t_query -----------------------------------
__global__ __launch_bounds__(256)
void g1_tquery(const float* __restrict__ Q, const float* __restrict__ W,
               ushort_t* __restrict__ tq) {
  __shared__ ushort_t As[2][BM][LDS_STRIDE];
  __shared__ ushort_t Bs[2][BN][LDS_STRIDE];
  const int t = threadIdx.x, lane = t & 31, wave = t >> 5;
  const int wm = wave & 3, wn = wave >> 2;
  const int m0 = blockIdx.y * BM, n0 = blockIdx.x * BN;

  v8f acc[2][4] = {};

  auto stage = [&](int kt, int buf) {
    const int k0 = kt * BK;
#pragma unroll
    for (int p = 0; p < 4; ++p) {                 // f32 -> bf16 convert staging
      const int f = t + p * 256, row = f >> 3, kq = (f & 7) * 4;
      const float4 a = *(const float4*)(Q + (size_t)(m0 + row) * D_DIM + k0 + kq);
      *(uint2*)&As[buf][row][kq] = make_uint2(pack2(a.x, a.y), pack2(a.z, a.w));
      const float4 b = *(const float4*)(W + (size_t)(n0 + row) * D_DIM + k0 + kq);
      *(uint2*)&Bs[buf][row][kq] = make_uint2(pack2(b.x, b.y), pack2(b.z, b.w));
    }
  };

  stage(0, 0);
  int buf = 0;
  for (int kt = 0; kt < NKC; ++kt) {              // double-buffered pipeline
    __syncthreads();
    if (kt + 1 < NKC) stage(kt + 1, buf ^ 1);
    mma_tiles(As[buf], Bs[buf], wm, wn, lane, acc);
    buf ^= 1;
  }

  const int colL = lane & 15, rhi = (lane >> 4) * 8;
#pragma unroll
  for (int mt = 0; mt < 2; ++mt)
#pragma unroll
    for (int nt = 0; nt < 4; ++nt)
#pragma unroll
      for (int j = 0; j < 8; ++j) {
        const int m = m0 + wm * 32 + mt * 16 + rhi + j;
        const int n = n0 + wn * 64 + nt * 16 + colL;
        tq[(size_t)m * D_DIM + n] = f2bf(acc[mt][nt][j]);
      }
}

// ------------------------- GEMM2: alpha (masked) -----------------------------
__global__ __launch_bounds__(256)
void g2_alpha(const ushort_t* __restrict__ tq, const ushort_t* __restrict__ keysN,
              const unsigned char* __restrict__ Mask, float* __restrict__ alphaOut) {
  __shared__ ushort_t As[2][BM][LDS_STRIDE];
  __shared__ ushort_t Bs[2][BN][LDS_STRIDE];
  const int t = threadIdx.x, lane = t & 31, wave = t >> 5;
  const int wm = wave & 3, wn = wave >> 2;
  const int m0 = blockIdx.y * BM, n0 = blockIdx.x * BN;  // m0: s-tile, n0: k-tile
  const int b = blockIdx.z;

  v8f acc[2][4] = {};

  auto stage = [&](int kt, int buf) {             // raw bf16 copies -> async path
    const int k0 = kt * BK;
#pragma unroll
    for (int p = 0; p < 2; ++p) {
      const int f = t + p * 256, row = f >> 2, kq = (f & 3) * 8;
      async_copy16(tq + ((size_t)(m0 + row) * B_DIM + b) * D_DIM + k0 + kq,
                   &As[buf][row][kq]);
      async_copy16(keysN + ((size_t)b * K_DIM + n0 + row) * D_DIM + k0 + kq,
                   &Bs[buf][row][kq]);
    }
  };

  stage(0, 0);
  int buf = 0;
  for (int kt = 0; kt < NKC; ++kt) {
    async_wait();
    __syncthreads();
    if (kt + 1 < NKC) stage(kt + 1, buf ^ 1);
    mma_tiles(As[buf], Bs[buf], wm, wn, lane, acc);
    buf ^= 1;
  }

  const int colL = lane & 15, rhi = (lane >> 4) * 8;
  const float ninf = -__builtin_inff();
#pragma unroll
  for (int mt = 0; mt < 2; ++mt)
#pragma unroll
    for (int nt = 0; nt < 4; ++nt)
#pragma unroll
      for (int j = 0; j < 8; ++j) {
        const int s = m0 + wm * 32 + mt * 16 + rhi + j;
        const int k = n0 + wn * 64 + nt * 16 + colL;
        float v = acc[mt][nt][j];
        if (Mask[b * K_DIM + k]) v = ninf;
        alphaOut[((size_t)s * B_DIM + b) * K_DIM + k] = v;
      }
}

// ------------------------- softmax row stats ---------------------------------
__global__ __launch_bounds__(256)
void g_softmax_stats(const float* __restrict__ alpha, float* __restrict__ rowmax,
                     float* __restrict__ rowinv) {
  const int r = blockIdx.x * 8 + (threadIdx.x >> 5);
  const int lane = threadIdx.x & 31;
  const float* row = alpha + (size_t)r * K_DIM;
  float vals[K_DIM / 32];
  float m = -__builtin_inff();
#pragma unroll
  for (int i = 0; i < K_DIM / 32; ++i) {
    vals[i] = row[lane + 32 * i];
    m = fmaxf(m, vals[i]);
  }
#pragma unroll
  for (int off = 16; off > 0; off >>= 1) m = fmaxf(m, __shfl_xor(m, off, 32));
  float s = 0.f;
#pragma unroll
  for (int i = 0; i < K_DIM / 32; ++i) s += __expf(vals[i] - m);
#pragma unroll
  for (int off = 16; off > 0; off >>= 1) s += __shfl_xor(s, off, 32);
  if (lane == 0) { rowmax[r] = m; rowinv[r] = 1.0f / s; }
}

// ------------------------- GEMM3: attened ------------------------------------
__global__ __launch_bounds__(256)
void g3_attend(const float* __restrict__ alphaIn, const ushort_t* __restrict__ keysT,
               const float* __restrict__ rowmax, const float* __restrict__ rowinv,
               float* __restrict__ attOut) {
  __shared__ ushort_t As[2][BM][LDS_STRIDE];
  __shared__ ushort_t Bs[2][BN][LDS_STRIDE];
  const int t = threadIdx.x, lane = t & 31, wave = t >> 5;
  const int wm = wave & 3, wn = wave >> 2;
  const int m0 = blockIdx.y * BM, n0 = blockIdx.x * BN;  // m0: s-tile, n0: d-tile
  const int b = blockIdx.z;

  v8f acc[2][4] = {};

  auto stage = [&](int kt, int buf) {
    const int k0 = kt * BK;
#pragma unroll
    for (int p = 0; p < 4; ++p) {   // A: weights = exp(alpha - max) * invsum
      const int f = t + p * 256, row = f >> 3, kq = (f & 7) * 4;
      const size_t r = (size_t)(m0 + row) * B_DIM + b;
      const float mx = rowmax[r], iv = rowinv[r];
      const float4 a = *(const float4*)(alphaIn + r * K_DIM + k0 + kq);
      const float w0 = __expf(a.x - mx) * iv, w1 = __expf(a.y - mx) * iv;
      const float w2 = __expf(a.z - mx) * iv, w3 = __expf(a.w - mx) * iv;
      *(uint2*)&As[buf][row][kq] = make_uint2(pack2(w0, w1), pack2(w2, w3));
    }
#pragma unroll
    for (int p = 0; p < 2; ++p) {   // B: raw bf16 rows of keysT -> async path
      const int f = t + p * 256, row = f >> 2, kq = (f & 3) * 8;
      async_copy16(keysT + ((size_t)b * D_DIM + n0 + row) * K_DIM + k0 + kq,
                   &Bs[buf][row][kq]);
    }
  };

  stage(0, 0);
  int buf = 0;
  for (int kt = 0; kt < NKC; ++kt) {
    async_wait();
    __syncthreads();
    if (kt + 1 < NKC) stage(kt + 1, buf ^ 1);
    mma_tiles(As[buf], Bs[buf], wm, wn, lane, acc);
    buf ^= 1;
  }

  const int colL = lane & 15, rhi = (lane >> 4) * 8;
#pragma unroll
  for (int mt = 0; mt < 2; ++mt)
#pragma unroll
    for (int nt = 0; nt < 4; ++nt)
#pragma unroll
      for (int j = 0; j < 8; ++j) {
        const int s = m0 + wm * 32 + mt * 16 + rhi + j;
        const int d = n0 + wn * 64 + nt * 16 + colL;
        attOut[((size_t)s * B_DIM + b) * D_DIM + d] = acc[mt][nt][j];
      }
}

// ---------------------------------------------------------------------------
extern "C" void kernel_launch(void* const* d_in, const int* in_sizes, int n_in,
                              void* d_out, int out_size, void* d_ws, size_t ws_size,
                              hipStream_t stream) {
  const float* Q            = (const float*)d_in[0];           // (S,B,QD) f32
  const float* Keys         = (const float*)d_in[1];           // (B,K,KD) f32
  const unsigned char* Mask = (const unsigned char*)d_in[2];   // (B,K) bool, 1 byte
  const float* W            = (const float*)d_in[3];           // (KD,QD) f32

  float* out      = (float*)d_out;
  float* attOut   = out;                                       // (S,B,KD) first
  float* alphaOut = out + (size_t)M1 * K_DIM;                  // (S,B,K) second

  // ws: tq bf16 32MiB | keysN bf16 32MiB | keysT bf16 32MiB | rowmax | rowinv
  ushort_t* tq    = (ushort_t*)d_ws;
  ushort_t* keysN = tq + (size_t)M1 * D_DIM;
  ushort_t* keysT = keysN + (size_t)B_DIM * K_DIM * D_DIM;
  float* rowmax   = (float*)(keysT + (size_t)B_DIM * D_DIM * K_DIM);
  float* rowinv   = rowmax + M1;

  const dim3 blk(256);
  k0_convert<<<dim3(D_DIM / 32, K_DIM / 32, B_DIM), blk, 0, stream>>>(Keys, keysN, keysT);
  g1_tquery<<<dim3(D_DIM / BN, M1 / BM), blk, 0, stream>>>(Q, W, tq);
  g2_alpha<<<dim3(K_DIM / BN, S_DIM / BM, B_DIM), blk, 0, stream>>>(tq, keysN, Mask, alphaOut);
  g_softmax_stats<<<dim3(M1 / 8), blk, 0, stream>>>(alphaOut, rowmax, rowinv);
  g3_attend<<<dim3(D_DIM / BN, S_DIM / BM, B_DIM), blk, 0, stream>>>(alphaOut, keysT, rowmax, rowinv, attOut);
}
